// NNBase_94489280702
// MI455X (gfx1250) — matrix operand order repaired and checked by reading, not compiled, not run on hardware
//
#include <hip/hip_runtime.h>
#include <hip/hip_bf16.h>

// Problem sizes (fixed by the reference)
#define T_STEPS 1024
#define N_BATCH 32
#define IN_DIM  512
#define H_DIM   512
#define G3      1536          // 3*H
#define KT      16            // K tiles of 32 (512/32) for bf16 WMMA
#define CT_TOT  96            // column tiles of 16 (1536/16)

// Scan configuration: 16 workgroups, each owns 32 h-columns (and the aligned
// r/z/n gate columns), keeps its 96KB w_hh slice resident in LDS.
#define NBLK         16
#define SCAN_THREADS 384      // 12 wave32 -> 12 output tiles (2 M x 6 N) per WG
#define HC           32
#define NH           (N_BATCH * H_DIM)   // 16384

typedef __attribute__((ext_vector_type(16))) __bf16       v16bf;
typedef __attribute__((ext_vector_type(8)))  __bf16       v8bf;
typedef __attribute__((ext_vector_type(8)))  float        v8f;
typedef __attribute__((ext_vector_type(4)))  unsigned int u32x4;
typedef __attribute__((ext_vector_type(8)))  unsigned int u32x8;

// ---- workspace layout (bytes). Requires ws_size >= ~104 MB. ----
#define GI_BYTES   ((size_t)T_STEPS * N_BATCH * G3 * 2)        // 100663296
#define WP_BYTES   ((size_t)G3 * IN_DIM * 2)                   // 1572864
#define HX_BYTES   ((size_t)2 * N_BATCH * H_DIM * 4)           // 131072
#define SMEM_BYTES (6*16*512*2 + NH*2 + NH*4 + N_BATCH*96*4)   // 208896

static __device__ __forceinline__ v16bf cat8(v8bf a, v8bf b) {
  return __builtin_shufflevector(a, b, 0,1,2,3,4,5,6,7,8,9,10,11,12,13,14,15);
}

// ---------------------------------------------------------------------------
// Tensor Data Mover: 1D contiguous global->LDS DMA (dword-granular).
// D# group0 (4 SGPRs): count=1 | lds_addr | global_addr[31:0] |
//                      global_addr[56:32] + type=2 in [127:126].
// D# group1 (8 SGPRs): data_size=4B, tensor_dim0=tile_dim0=stride=ndwords,
//                      no multicast / iterate / pad / gather.
// Issued once per wave (EXEC-independent); tracked with TENSORcnt.
// ---------------------------------------------------------------------------
static __device__ __forceinline__ void tdm_load_1d(unsigned lds_byte_off,
                                                   const void* gsrc,
                                                   unsigned ndwords) {
  const unsigned long long ga = (unsigned long long)(size_t)gsrc;
  u32x4 g0;
  g0[0] = 1u;                                            // count=1 (valid user D#)
  g0[1] = lds_byte_off;                                  // lds_addr (bytes)
  g0[2] = (unsigned)(ga & 0xffffffffu);                  // global_addr[31:0]
  g0[3] = (unsigned)((ga >> 32) & 0x01ffffffu) | (2u << 30); // addr[56:32] | type=2
  u32x8 g1;
  g1[0] = (2u << 16);                                    // data_size=2 (4 bytes)
  g1[1] = (ndwords & 0xffffu) << 16;                     // tensor_dim0[15:0]
  g1[2] = (ndwords >> 16) & 0xffffu;                     // tensor_dim0[31:16]
  g1[3] = (ndwords & 0xffffu) << 16;                     // tile_dim0[15:0]
  g1[4] = 0u;                                            // tile_dim1/2 unused
  g1[5] = ndwords;                                       // tensor_dim0_stride[31:0]
  g1[6] = 0u;
  g1[7] = 0u;
  asm volatile("tensor_load_to_lds %0, %1" :: "s"(g0), "s"(g1) : "memory");
}

static __device__ __forceinline__ unsigned lds_off_of(const void* p) {
  // Flat->LDS mapping keeps the LDS offset in addr[31:0].
  return (unsigned)(size_t)p;
}

// Pack weight w [1536][512] f32 into bf16 B-tile layout:
// packed[(ct*KT + kt)*512 + lane*16 + j] = w[ct*16 + (lane&15)][kt*32 + 2*(j&8) + 8*(lane>>4) + (j&7)]
// (lane l holds column N=l&15 of the 32x16 B tile, 16 bf16 in table K order).
__global__ void __launch_bounds__(256)
pack_weights_kernel(const float* __restrict__ w_ih, const float* __restrict__ w_hh,
                    __bf16* __restrict__ p_ih, __bf16* __restrict__ p_hh,
                    unsigned* __restrict__ barrier_cnt) {
  if (blockIdx.x == 0 && threadIdx.x == 0) *barrier_cnt = 0u;  // reset per launch
  const int PER_W = CT_TOT * KT * 512;                          // 786432
  int p = blockIdx.x * blockDim.x + threadIdx.x;
  if (p >= 2 * PER_W) return;
  const float* w  = (p < PER_W) ? w_ih : w_hh;
  __bf16*      dst = (p < PER_W) ? p_ih : p_hh;
  int q    = (p < PER_W) ? p : p - PER_W;
  int ct   = q / (KT * 512);
  int rem  = q % (KT * 512);
  int kt   = rem / 512;
  int e    = rem % 512;
  int lane = e >> 4;
  int j    = e & 15;
  dst[q] = (__bf16)w[(ct * 16 + (lane & 15)) * IN_DIM + (kt * 32 + ((j & 8) << 1) + ((lane >> 4) << 3) + (j & 7))];
}

// gi = x @ w_ih^T + b_ih, stored bf16. One wave computes a 16x64 slab
// (A fragment reused across 4 B tiles per K step).
__global__ void __launch_bounds__(256)
gi_gemm_kernel(const float* __restrict__ x, const __bf16* __restrict__ p_ih,
               const float* __restrict__ b_ih, __bf16* __restrict__ gi) {
  const int lane  = threadIdx.x & 31;
  const int wave  = threadIdx.x >> 5;
  const int gid   = blockIdx.x * 8 + wave;     // 49152 wave tasks
  const int mtile = gid / 24;                  // 2048 M tiles
  const int cg    = gid % 24;                  // 24 groups of 4 column tiles
  const int m     = mtile * 16 + (lane & 15);
  const int ksel  = (lane >> 4) * 8;           // A layout K selector

  const v8f zero = {};
  v8f c[4];
#pragma unroll
  for (int i = 0; i < 4; ++i) c[i] = zero;

  const float* xrow = x + (size_t)m * IN_DIM;
  for (int kt = 0; kt < KT; ++kt) {
    const int k0 = kt * 32;
    v8f lo = *(const v8f*)(xrow + k0 + ksel);
    v8f hi = *(const v8f*)(xrow + k0 + 16 + ksel);
    v16bf a;
#pragma unroll
    for (int i = 0; i < 8; ++i) { a[i] = (__bf16)lo[i]; a[8 + i] = (__bf16)hi[i]; }
#pragma unroll
    for (int i = 0; i < 4; ++i) {
      const __bf16* bp = p_ih + ((size_t)(cg * 4 + i) * KT + kt) * 512 + lane * 16;
      v8bf b0 = *(const v8bf*)bp;
      v8bf b1 = *(const v8bf*)(bp + 8);
      c[i] = __builtin_amdgcn_wmma_f32_16x16x32_bf16(
          false, a, false, cat8(b0, b1), (short)0, c[i], false, false);
    }
    if (kt + 1 < KT) __builtin_prefetch(xrow + (kt + 1) * 32 + ksel, 0, 3);
  }
  // C/D layout: VGPR v, lane l -> (M = v + 8*(l>>4), N = l&15)
  const int mrow = mtile * 16 + 8 * (lane >> 4);
#pragma unroll
  for (int i = 0; i < 4; ++i) {
    const int col  = (cg * 4 + i) * 16 + (lane & 15);
    const float bb = b_ih[col];
#pragma unroll
    for (int v = 0; v < 8; ++v)
      gi[(size_t)(mrow + v) * G3 + col] = (__bf16)(c[i][v] + bb);
  }
}

// One-barrier-per-step grid sync over a monotonic counter in L2.
static __device__ __forceinline__ void grid_barrier(unsigned* cnt, unsigned target) {
  __syncthreads();
  if (threadIdx.x == 0) {
    __threadfence();  // release h_new stores
    __hip_atomic_fetch_add(cnt, 1u, __ATOMIC_RELEASE, __HIP_MEMORY_SCOPE_AGENT);
    while (__hip_atomic_load(cnt, __ATOMIC_ACQUIRE, __HIP_MEMORY_SCOPE_AGENT) < target)
      __builtin_amdgcn_s_sleep(1);
    __threadfence();  // acquire
  }
  __syncthreads();
}

// Persistent masked-GRU scan. WG b owns h columns [32b, 32b+32) and gate
// columns {j, 512+j, 1024+j} so it can update its h slice independently.
// w_hh slice and per-step hx are pulled into LDS by the Tensor Data Mover.
__global__ void __launch_bounds__(SCAN_THREADS)
gru_scan_kernel(const float* __restrict__ hxs, const float* __restrict__ masks,
                const __bf16* __restrict__ gi, const __bf16* __restrict__ p_hh,
                const float* __restrict__ b_hh, float* __restrict__ out,
                float* __restrict__ hx_pp, unsigned* __restrict__ bar) {
  extern __shared__ __align__(16) char smem[];
  __bf16* lds_w   = (__bf16*)smem;                                  // 98304 B
  __bf16* lds_hxb = (__bf16*)(smem + 6*16*512*2);                   // 32768 B
  float*  lds_hxf = (float*)(smem + 6*16*512*2 + NH*2);             // 65536 B
  float*  lds_gh  = (float*)(smem + 6*16*512*2 + NH*2 + NH*4);      // 12288 B

  const int tid  = threadIdx.x;
  const int lane = tid & 31;
  const int wave = tid >> 5;       // 0..11
  const int blk  = blockIdx.x;     // 0..15
  const int hc0  = blk * HC;

  // ---- TDM: stage this WG's w_hh slice (6 x 16KB tiles) into LDS ----
  if (wave == 0) {
#pragma unroll
    for (int lt = 0; lt < 6; ++lt) {
      const int ct = (lt >> 1) * 32 + 2 * blk + (lt & 1);
      tdm_load_1d(lds_off_of(lds_w + lt * 16 * 512),
                  p_hh + (size_t)ct * 16 * 512, 16 * 512 / 2 /*dwords*/);
    }
    __builtin_amdgcn_s_wait_tensorcnt(0);
  }
  __syncthreads();

  const int mt   = wave & 1;       // M tile (rows 0-15 / 16-31)
  const int lt   = wave >> 1;      // local column tile 0..5
  const int m_a  = mt * 16 + (lane & 15);
  const int ksel = (lane >> 4) * 8;

  for (int t = 0; t < T_STEPS; ++t) {
    // ---- TDM: pull this step's hidden state (64KB f32) into LDS ----
    if (wave == 0) {
      const float* hsrc = (t == 0) ? hxs : (hx_pp + (size_t)(t & 1) * NH);
      tdm_load_1d(lds_off_of(lds_hxf), hsrc, NH /*dwords*/);
      __builtin_amdgcn_s_wait_tensorcnt(0);
    }
    __syncthreads();

    // Apply step mask; keep f32 (for z*hx) and bf16 (WMMA A) copies.
    for (int e = tid; e < NH; e += SCAN_THREADS) {
      const int n = e >> 9;
      const float v = lds_hxf[e] * masks[t * N_BATCH + n];
      lds_hxf[e] = v;
      lds_hxb[e] = (__bf16)v;
    }
    __syncthreads();

    // gh tile (16x16) per wave: hx_masked @ w_hh^T, K=512 via 16 bf16 WMMAs.
    v8f acc = {};
    for (int kt = 0; kt < KT; ++kt) {
      const __bf16* ap = lds_hxb + m_a * H_DIM + kt * 32 + ksel;
      v8bf a0 = *(const v8bf*)ap;
      v8bf a1 = *(const v8bf*)(ap + 16);
      const __bf16* bp = lds_w + (lt * 16 + kt) * 512 + lane * 16;
      v8bf b0 = *(const v8bf*)bp;
      v8bf b1 = *(const v8bf*)(bp + 8);
      acc = __builtin_amdgcn_wmma_f32_16x16x32_bf16(
          false, cat8(a0, a1), false, cat8(b0, b1), (short)0, acc, false, false);
    }
    {
      const int row0 = mt * 16 + 8 * (lane >> 4);
      const int coll = lt * 16 + (lane & 15);
#pragma unroll
      for (int v = 0; v < 8; ++v) lds_gh[(row0 + v) * 96 + coll] = acc[v];
    }
    __syncthreads();

    // Gate math + state update for this WG's 32 h columns.
    for (int e = tid; e < N_BATCH * HC; e += SCAN_THREADS) {
      const int n  = e >> 5;
      const int jc = e & 31;
      const int j  = hc0 + jc;
      const __bf16* girow = gi + (size_t)(t * N_BATCH + n) * G3;
      const float ir  = (float)girow[j];
      const float iz  = (float)girow[H_DIM + j];
      const float inn = (float)girow[2 * H_DIM + j];
      const float hr  = lds_gh[n * 96 + jc]      + b_hh[j];
      const float hz  = lds_gh[n * 96 + 32 + jc] + b_hh[H_DIM + j];
      const float hn  = lds_gh[n * 96 + 64 + jc] + b_hh[2 * H_DIM + j];
      const float hxm = lds_hxf[n * H_DIM + j];
      const float r  = 1.f / (1.f + __expf(-(ir + hr)));
      const float z  = 1.f / (1.f + __expf(-(iz + hz)));
      const float ng = tanhf(inn + r * hn);
      const float hnew = (1.f - z) * ng + z * hxm;
      out[(size_t)(t * N_BATCH + n) * H_DIM + j] = hnew;
      hx_pp[(size_t)((t + 1) & 1) * NH + n * H_DIM + j] = hnew;
      if (t == T_STEPS - 1)
        out[(size_t)T_STEPS * NH + n * H_DIM + j] = hnew;
    }

    grid_barrier(bar, (unsigned)(t + 1) * NBLK);
  }
}

extern "C" void kernel_launch(void* const* d_in, const int* in_sizes, int n_in,
                              void* d_out, int out_size, void* d_ws, size_t ws_size,
                              hipStream_t stream) {
  const float* x     = (const float*)d_in[0];
  const float* hxs   = (const float*)d_in[1];
  const float* masks = (const float*)d_in[2];
  const float* w_ih  = (const float*)d_in[3];
  const float* w_hh  = (const float*)d_in[4];
  const float* b_ih  = (const float*)d_in[5];
  const float* b_hh  = (const float*)d_in[6];
  float* out = (float*)d_out;

  char* ws = (char*)d_ws;   // requires ws_size >= ~104 MB (see layout above)
  __bf16*   gi   = (__bf16*)ws;
  __bf16*   pih  = (__bf16*)(ws + GI_BYTES);
  __bf16*   phh  = (__bf16*)(ws + GI_BYTES + WP_BYTES);
  float*    hxpp = (float*)(ws + GI_BYTES + 2 * WP_BYTES);
  unsigned* bar  = (unsigned*)(ws + GI_BYTES + 2 * WP_BYTES + HX_BYTES);

  (void)in_sizes; (void)n_in; (void)out_size; (void)ws_size;

  hipFuncSetAttribute(reinterpret_cast<const void*>(gru_scan_kernel),
                      hipFuncAttributeMaxDynamicSharedMemorySize, SMEM_BYTES);

  pack_weights_kernel<<<6144, 256, 0, stream>>>(w_ih, w_hh, pih, phh, bar);
  gi_gemm_kernel<<<6144, 256, 0, stream>>>(x, pih, b_ih, gi);
  gru_scan_kernel<<<NBLK, SCAN_THREADS, SMEM_BYTES, stream>>>(
      hxs, masks, gi, phh, b_hh, out, hxpp, bar);
}
